// DecoderRNN_39513699123453
// MI455X (gfx1250) — matrix operand order repaired and checked by reading, not compiled, not run on hardware
//
#include <hip/hip_runtime.h>
#include <cstdint>
#include <cstddef>

// ---------------------------------------------------------------------------
// Shapes (match reference): B=32, P=196, E=256, A=512, M=512, D=512, V=20000,
// L=52, T=51
// ---------------------------------------------------------------------------
#define B_ 32
#define P_ 196
#define E_ 256
#define A_ 512
#define M_ 512
#define D_ 512
#define V_ 20000
#define L_ 52
#define T_ 51

typedef __attribute__((ext_vector_type(16))) __bf16 v16bf;
typedef __attribute__((ext_vector_type(8)))  float  v8f;

// float -> bf16 bits with round-to-nearest-even (bit trick; avoids relying on
// scalar __bf16 conversion support)
__device__ __forceinline__ unsigned short f2bf(float x) {
  unsigned int u = __float_as_uint(x);
  unsigned int r = 0x7FFFu + ((u >> 16) & 1u);
  u += r;
  return (unsigned short)(u >> 16);
}

__device__ __forceinline__ float sigmoidf_(float x) {
  return 1.0f / (1.0f + __expf(-x));
}

// ---------------------------------------------------------------------------
// WMMA-native packed layouts (per CDNA5 ISA, wave32):
// A tile 16x32 bf16: lane = (m&15) | (khalf<<4); khalf = (klocal>>3)&1;
//                    i = (klocal&7) | (((klocal>>4)&1)<<3)
// B tile 32x16 bf16: lane = (n&15) | (((klocal>>3)&1)<<4);
//                    i = (klocal&7) | (((klocal>>4)&1)<<3)
// Packed buffer: (((tile*KT)+kt)*32 + lane)*16 + i  -> each lane loads 16
// contiguous bf16 (32B) per k-step => global_load_b128 pairs.
// ---------------------------------------------------------------------------
__device__ __forceinline__ size_t a_pack_idx(int m, int k, int KT) {
  int mt = m >> 4, kt = k >> 5, kl = k & 31;
  int khalf = (kl >> 3) & 1;
  int i = (kl & 7) | (((kl >> 4) & 1) << 3);
  int lane = (m & 15) | (khalf << 4);
  return (((size_t)mt * KT + kt) * 32 + lane) * 16 + i;
}
__device__ __forceinline__ size_t b_pack_idx(int k, int n, int KT) {
  int nt = n >> 4, kt = k >> 5, kl = k & 31;
  int lane = (n & 15) | (((kl >> 3) & 1) << 4);
  int i = (kl & 7) | (((kl >> 4) & 1) << 3);
  return (((size_t)nt * KT + kt) * 32 + lane) * 16 + i;
}

// --------------------------- packing kernels -------------------------------
__global__ void pack_w_k(const float* __restrict__ src, unsigned short* __restrict__ dst,
                         int K, int N) {
  size_t idx = (size_t)blockIdx.x * blockDim.x + threadIdx.x;
  if (idx >= (size_t)K * N) return;
  int k = (int)(idx / N), n = (int)(idx % N);
  dst[b_pack_idx(k, n, K >> 5)] = f2bf(src[idx]);
}

__global__ void pack_a_k(const float* __restrict__ src, unsigned short* __restrict__ dst,
                         int M, int K) {
  size_t idx = (size_t)blockIdx.x * blockDim.x + threadIdx.x;
  if (idx >= (size_t)M * K) return;
  int m = (int)(idx / K), k = (int)(idx % K);
  dst[a_pack_idx(m, k, K >> 5)] = f2bf(src[idx]);
}

// Gather embeddings for all (t,b) and pack into per-step A buffers (K=768,
// embeddings occupy k=0..511; awe fills k=512..767 each step).
__global__ void embed_pack_k(const int* __restrict__ caps, const float* __restrict__ emb,
                             unsigned short* __restrict__ pAih) {
  size_t idx = (size_t)blockIdx.x * blockDim.x + threadIdx.x;
  if (idx >= (size_t)T_ * B_ * M_) return;
  int t = (int)(idx / (B_ * M_));
  int r = (int)(idx % (B_ * M_));
  int b = r / M_, k = r % M_;
  int tok = caps[b * L_ + t];
  unsigned short* dst = pAih + (size_t)t * (2 * 24 * 512);
  dst[a_pack_idx(b, k, 24)] = f2bf(emb[(size_t)tok * M_ + k]);
}

__global__ void pack_awe_k(const float* __restrict__ aweG, unsigned short* __restrict__ pAih_t) {
  int idx = blockIdx.x * blockDim.x + threadIdx.x;
  if (idx >= B_ * E_) return;
  int b = idx / E_, e = idx % E_;
  pAih_t[a_pack_idx(b, M_ + e, 24)] = f2bf(aweG[idx]);
}

// --------------------------- generic WMMA GEMM -----------------------------
// One wave computes a 16x16 fp32 tile: D = A(bf16) x B(bf16) + bias, optional
// sigmoid. A: Mtiles x KT packed; B: Ntiles x KT packed. Out row-major MxN.
__global__ void gemm_bias_k(const unsigned short* __restrict__ pA,
                            const unsigned short* __restrict__ pB,
                            const float* __restrict__ bias, float* __restrict__ out,
                            int Mtiles, int Ntiles, int KT, int N, int act) {
  int wave = blockIdx.x * (blockDim.x >> 5) + (threadIdx.x >> 5);
  if (wave >= Mtiles * Ntiles) return;   // wave-uniform: EXEC stays all-ones
  int lane = threadIdx.x & 31;
  int mt = wave % Mtiles, nt = wave / Mtiles;
  const unsigned short* pa = pA + (size_t)mt * KT * 512 + lane * 16;
  const unsigned short* pb = pB + (size_t)nt * KT * 512 + lane * 16;
  v8f c = {};
  for (int kt = 0; kt < KT; ++kt) {
    v16bf a = *(const v16bf*)pa; pa += 512;
    v16bf b = *(const v16bf*)pb; pb += 512;
    c = __builtin_amdgcn_wmma_f32_16x16x32_bf16(false, a, false, b, (short)0, c, false, false);
  }
  int row0 = mt * 16 + ((lane >> 4) << 3);
  int col  = nt * 16 + (lane & 15);
  float bv = bias[col];
#pragma unroll
  for (int v = 0; v < 8; ++v) {
    float x = c[v] + bv;
    if (act == 1) x = sigmoidf_(x);
    out[(size_t)(row0 + v) * N + col] = x;
  }
}

// gates = [e_t, awe](32x768) @ W_ih + b_ih + h(32x512) @ W_hh + b_hh
__global__ void gemm_gates_k(const unsigned short* __restrict__ pA1,
                             const unsigned short* __restrict__ pW1,
                             const float* __restrict__ b1,
                             const unsigned short* __restrict__ pA2,
                             const unsigned short* __restrict__ pW2,
                             const float* __restrict__ b2,
                             float* __restrict__ out) {
  int wave = blockIdx.x * (blockDim.x >> 5) + (threadIdx.x >> 5);
  if (wave >= 2 * 128) return;
  int lane = threadIdx.x & 31;
  int mt = wave & 1, nt = wave >> 1;
  v8f c = {};
  const unsigned short* pa = pA1 + (size_t)mt * 24 * 512 + lane * 16;
  const unsigned short* pb = pW1 + (size_t)nt * 24 * 512 + lane * 16;
  for (int kt = 0; kt < 24; ++kt) {
    v16bf a = *(const v16bf*)pa; pa += 512;
    v16bf b = *(const v16bf*)pb; pb += 512;
    c = __builtin_amdgcn_wmma_f32_16x16x32_bf16(false, a, false, b, (short)0, c, false, false);
  }
  pa = pA2 + (size_t)mt * 16 * 512 + lane * 16;
  pb = pW2 + (size_t)nt * 16 * 512 + lane * 16;
  for (int kt = 0; kt < 16; ++kt) {
    v16bf a = *(const v16bf*)pa; pa += 512;
    v16bf b = *(const v16bf*)pb; pb += 512;
    c = __builtin_amdgcn_wmma_f32_16x16x32_bf16(false, a, false, b, (short)0, c, false, false);
  }
  int row0 = mt * 16 + ((lane >> 4) << 3);
  int col  = nt * 16 + (lane & 15);
  float bv = b1[col] + b2[col];
#pragma unroll
  for (int v = 0; v < 8; ++v)
    out[(size_t)(row0 + v) * (4 * D_) + col] = c[v] + bv;
}

// preds[b,t,:] = mask(b,t) ? h @ W_fc + b_fc : 0   (32x512 @ 512x20000)
__global__ void gemm_preds_k(const unsigned short* __restrict__ pH,
                             const unsigned short* __restrict__ pWfc,
                             const float* __restrict__ bfc,
                             const int* __restrict__ cap_len,
                             float* __restrict__ preds, int t) {
  int wave = blockIdx.x * (blockDim.x >> 5) + (threadIdx.x >> 5);
  if (wave >= 2 * (V_ / 16)) return;
  int lane = threadIdx.x & 31;
  int mt = wave & 1, nt = wave >> 1;
  v8f c = {};
  const unsigned short* pa = pH  + (size_t)mt * 16 * 512 + lane * 16;
  const unsigned short* pb = pWfc + (size_t)nt * 16 * 512 + lane * 16;
  for (int kt = 0; kt < 16; ++kt) {
    v16bf a = *(const v16bf*)pa; pa += 512;
    v16bf b = *(const v16bf*)pb; pb += 512;
    c = __builtin_amdgcn_wmma_f32_16x16x32_bf16(false, a, false, b, (short)0, c, false, false);
  }
  int row0 = mt * 16 + ((lane >> 4) << 3);
  int col  = nt * 16 + (lane & 15);
  float bv = bfc[col];
#pragma unroll
  for (int v = 0; v < 8; ++v) {
    int b = row0 + v;
    bool m = t < (cap_len[b] - 1);
    preds[((size_t)b * T_ + t) * V_ + col] = m ? (c[v] + bv) : 0.0f;
  }
}

// --------------------------- small fp32 kernels ----------------------------
__global__ void mean_enc_k(const float* __restrict__ enc, float* __restrict__ mean) {
  int idx = blockIdx.x * blockDim.x + threadIdx.x;
  if (idx >= B_ * E_) return;
  int b = idx / E_, e = idx % E_;
  const float* pe = enc + (size_t)b * P_ * E_ + e;
  float s = 0.0f;
  for (int p = 0; p < P_; ++p) s += pe[(size_t)p * E_];
  mean[idx] = s * (1.0f / (float)P_);
}

// att[b,p] = b_full + sum_a relu(att1[b,p,a] + att2[b,a]) * w_full[a]
__global__ void att_score_k(const float* __restrict__ att1, const float* __restrict__ att2,
                            const float* __restrict__ wfull, const float* __restrict__ bfull,
                            float* __restrict__ att) {
  int wave = blockIdx.x * (blockDim.x >> 5) + (threadIdx.x >> 5);
  if (wave >= B_ * P_) return;
  int lane = threadIdx.x & 31;
  int b = wave / P_;
  const float* r1 = att1 + (size_t)wave * A_;
  const float* r2 = att2 + (size_t)b * A_;
  float s = 0.0f;
  for (int a = lane; a < A_; a += 32) {
    float v = r1[a] + r2[a];
    v = v > 0.0f ? v : 0.0f;
    s += v * wfull[a];
  }
#pragma unroll
  for (int off = 16; off > 0; off >>= 1) s += __shfl_xor(s, off, 32);
  if (lane == 0) att[wave] = s + bfull[0];
}

// per-batch softmax over P, masked alpha output, gated attention-weighted enc
__global__ void softmax_awe_k(const float* __restrict__ att, const float* __restrict__ enc,
                              const float* __restrict__ gate, float* __restrict__ aweG,
                              float* __restrict__ alphas_out,
                              const int* __restrict__ cap_len, int t) {
  __shared__ float sh[256];
  __shared__ float alpha[P_];
  int b = blockIdx.x, tid = threadIdx.x;
  float v = (tid < P_) ? att[b * P_ + tid] : -1e30f;
  sh[tid] = v; __syncthreads();
  for (int s = 128; s > 0; s >>= 1) { if (tid < s) sh[tid] = fmaxf(sh[tid], sh[tid + s]); __syncthreads(); }
  float mx = sh[0]; __syncthreads();
  float e = (tid < P_) ? __expf(v - mx) : 0.0f;
  sh[tid] = e; __syncthreads();
  for (int s = 128; s > 0; s >>= 1) { if (tid < s) sh[tid] += sh[tid + s]; __syncthreads(); }
  float inv = 1.0f / sh[0];
  bool m = t < (cap_len[b] - 1);
  if (tid < P_) {
    float al = e * inv;
    alpha[tid] = al;
    alphas_out[((size_t)b * T_ + t) * P_ + tid] = m ? al : 0.0f;
  }
  __syncthreads();
  // tid indexes the E=256 channel
  float acc = 0.0f;
  const float* encb = enc + (size_t)b * P_ * E_;
  for (int p = 0; p < P_; ++p) acc += alpha[p] * encb[(size_t)p * E_ + tid];
  aweG[b * E_ + tid] = acc * gate[b * E_ + tid];
}

// LSTM cell + masked state update (h,c updated only where t < len-1)
__global__ void lstm_k(const float* __restrict__ gates, float* __restrict__ h,
                       float* __restrict__ c, const int* __restrict__ cap_len, int t) {
  int idx = blockIdx.x * blockDim.x + threadIdx.x;
  if (idx >= B_ * D_) return;
  int b = idx >> 9, d = idx & 511;
  const float* g = gates + (size_t)b * (4 * D_);
  float gi = sigmoidf_(g[d]);
  float gf = sigmoidf_(g[D_ + d]);
  float gg = tanhf(g[2 * D_ + d]);
  float go = sigmoidf_(g[3 * D_ + d]);
  float cn = gf * c[idx] + gi * gg;
  float hn = go * tanhf(cn);
  if (t < (cap_len[b] - 1)) { c[idx] = cn; h[idx] = hn; }
}

__global__ void copyout_k(const int* __restrict__ caps, const int* __restrict__ cap_len,
                          float* __restrict__ out_caps, float* __restrict__ out_dl) {
  int idx = blockIdx.x * blockDim.x + threadIdx.x;
  if (idx < B_ * L_) out_caps[idx] = (float)caps[idx];
  if (idx < B_) out_dl[idx] = (float)(cap_len[idx] - 1);
}

// ---------------------------------------------------------------------------
extern "C" void kernel_launch(void* const* d_in, const int* in_sizes, int n_in,
                              void* d_out, int out_size, void* d_ws, size_t ws_size,
                              hipStream_t stream) {
  (void)in_sizes; (void)n_in; (void)out_size; (void)ws_size;
  const float* enc      = (const float*)d_in[0];
  const int*   caps     = (const int*)d_in[1];
  const int*   cap_len  = (const int*)d_in[2];
  const float* W_enc    = (const float*)d_in[3];
  const float* b_enc    = (const float*)d_in[4];
  const float* W_dec    = (const float*)d_in[5];
  const float* b_dec    = (const float*)d_in[6];
  const float* W_full   = (const float*)d_in[7];
  const float* b_full   = (const float*)d_in[8];
  const float* emb      = (const float*)d_in[9];
  const float* W_ih     = (const float*)d_in[10];
  const float* b_ih     = (const float*)d_in[11];
  const float* W_hh     = (const float*)d_in[12];
  const float* b_hh     = (const float*)d_in[13];
  const float* W_init_h = (const float*)d_in[14];
  const float* b_init_h = (const float*)d_in[15];
  const float* W_init_c = (const float*)d_in[16];
  const float* b_init_c = (const float*)d_in[17];
  const float* W_fbeta  = (const float*)d_in[18];
  const float* b_fbeta  = (const float*)d_in[19];
  const float* W_fc     = (const float*)d_in[20];
  const float* b_fc     = (const float*)d_in[21];

  float* out        = (float*)d_out;
  float* out_preds  = out;                                   // 32*51*20000
  float* out_caps   = out + (size_t)B_ * T_ * V_;            // 32*52
  float* out_dl     = out_caps + (size_t)B_ * L_;            // 32
  float* out_alphas = out_dl + B_;                           // 32*51*196

  // ---- workspace carve (all offsets 256B aligned) ----
  uint8_t* wp = (uint8_t*)d_ws;
  auto alloc = [&](size_t bytes) -> void* {
    void* p = (void*)wp;
    wp += (bytes + 255) & ~(size_t)255;
    return p;
  };
  unsigned short* pWenc   = (unsigned short*)alloc((size_t)E_ * A_ * 2);
  unsigned short* pWih0   = (unsigned short*)alloc((size_t)E_ * D_ * 2);  // W_init_h
  unsigned short* pWic0   = (unsigned short*)alloc((size_t)E_ * D_ * 2);  // W_init_c
  unsigned short* pWdec   = (unsigned short*)alloc((size_t)D_ * A_ * 2);
  unsigned short* pWfb    = (unsigned short*)alloc((size_t)D_ * E_ * 2);
  unsigned short* pWih    = (unsigned short*)alloc((size_t)(M_ + E_) * 4 * D_ * 2);
  unsigned short* pWhh    = (unsigned short*)alloc((size_t)D_ * 4 * D_ * 2);
  unsigned short* pWfc    = (unsigned short*)alloc((size_t)D_ * V_ * 2);
  unsigned short* pEnc    = (unsigned short*)alloc((size_t)B_ * P_ * E_ * 2);
  unsigned short* pMean   = (unsigned short*)alloc((size_t)B_ * E_ * 2);
  unsigned short* pAih    = (unsigned short*)alloc((size_t)T_ * 2 * 24 * 512 * 2);
  unsigned short* pH      = (unsigned short*)alloc((size_t)B_ * D_ * 2);
  float* meanE  = (float*)alloc((size_t)B_ * E_ * 4);
  float* hBuf   = (float*)alloc((size_t)B_ * D_ * 4);
  float* cBuf   = (float*)alloc((size_t)B_ * D_ * 4);
  float* att1   = (float*)alloc((size_t)B_ * P_ * A_ * 4);
  float* att2   = (float*)alloc((size_t)B_ * A_ * 4);
  float* gateB  = (float*)alloc((size_t)B_ * E_ * 4);
  float* attS   = (float*)alloc((size_t)B_ * P_ * 4);
  float* aweG   = (float*)alloc((size_t)B_ * E_ * 4);
  float* gates  = (float*)alloc((size_t)B_ * 4 * D_ * 4);

  auto cdiv = [](long long a, long long b) { return (int)((a + b - 1) / b); };
  const int BS = 256;

  // ---- one-time precompute: bf16 pack of all weights into WMMA lane layout
  pack_w_k<<<cdiv((long long)E_ * A_, BS), BS, 0, stream>>>(W_enc, pWenc, E_, A_);
  pack_w_k<<<cdiv((long long)E_ * D_, BS), BS, 0, stream>>>(W_init_h, pWih0, E_, D_);
  pack_w_k<<<cdiv((long long)E_ * D_, BS), BS, 0, stream>>>(W_init_c, pWic0, E_, D_);
  pack_w_k<<<cdiv((long long)D_ * A_, BS), BS, 0, stream>>>(W_dec, pWdec, D_, A_);
  pack_w_k<<<cdiv((long long)D_ * E_, BS), BS, 0, stream>>>(W_fbeta, pWfb, D_, E_);
  pack_w_k<<<cdiv((long long)(M_ + E_) * 4 * D_, BS), BS, 0, stream>>>(W_ih, pWih, M_ + E_, 4 * D_);
  pack_w_k<<<cdiv((long long)D_ * 4 * D_, BS), BS, 0, stream>>>(W_hh, pWhh, D_, 4 * D_);
  pack_w_k<<<cdiv((long long)D_ * V_, BS), BS, 0, stream>>>(W_fc, pWfc, D_, V_);

  // encoder_out as packed A for att1; mean_enc; embeddings for every step
  pack_a_k<<<cdiv((long long)B_ * P_ * E_, BS), BS, 0, stream>>>(enc, pEnc, B_ * P_, E_);
  mean_enc_k<<<cdiv(B_ * E_, BS), BS, 0, stream>>>(enc, meanE);
  pack_a_k<<<cdiv(B_ * E_, BS), BS, 0, stream>>>(meanE, pMean, B_, E_);
  embed_pack_k<<<cdiv((long long)T_ * B_ * M_, BS), BS, 0, stream>>>(caps, emb, pAih);

  // h0 / c0 (32x256 @ 256x512), att1 (6272x256 @ 256x512)
  gemm_bias_k<<<cdiv(2 * 32, 8), BS, 0, stream>>>(pMean, pWih0, b_init_h, hBuf, 2, 32, 8, D_, 0);
  gemm_bias_k<<<cdiv(2 * 32, 8), BS, 0, stream>>>(pMean, pWic0, b_init_c, cBuf, 2, 32, 8, D_, 0);
  gemm_bias_k<<<cdiv(392 * 32, 8), BS, 0, stream>>>(pEnc, pWenc, b_enc, att1, 392, 32, 8, A_, 0);

  copyout_k<<<cdiv(B_ * L_, BS), BS, 0, stream>>>(caps, cap_len, out_caps, out_dl);

  // initial pack of h0
  pack_a_k<<<cdiv(B_ * D_, BS), BS, 0, stream>>>(hBuf, pH, B_, D_);

  // ---- sequential decode: 51 steps ----
  for (int t = 0; t < T_; ++t) {
    // att2 = h @ W_dec_att + b   (32x512 @ 512x512)
    gemm_bias_k<<<cdiv(2 * 32, 8), BS, 0, stream>>>(pH, pWdec, b_dec, att2, 2, 32, 16, A_, 0);
    // gate = sigmoid(h @ W_fbeta + b)   (32x512 @ 512x256)
    gemm_bias_k<<<cdiv(2 * 16, 8), BS, 0, stream>>>(pH, pWfb, b_fbeta, gateB, 2, 16, 16, E_, 1);
    // attention scores, softmax, gated weighted encoder sum
    att_score_k<<<cdiv(B_ * P_, 8), BS, 0, stream>>>(att1, att2, W_full, b_full, attS);
    softmax_awe_k<<<B_, BS, 0, stream>>>(attS, enc, gateB, aweG, out_alphas, cap_len, t);
    // place gated awe into this step's packed A (k = 512..767)
    pack_awe_k<<<cdiv(B_ * E_, BS), BS, 0, stream>>>(aweG, pAih + (size_t)t * (2 * 24 * 512));
    // gates = [e,awe]@W_ih + h@W_hh + biases   (32x2048 out)
    gemm_gates_k<<<cdiv(2 * 128, 8), BS, 0, stream>>>(
        pAih + (size_t)t * (2 * 24 * 512), pWih, b_ih, pH, pWhh, b_hh, gates);
    // LSTM cell + masked h/c update
    lstm_k<<<cdiv(B_ * D_, BS), BS, 0, stream>>>(gates, hBuf, cBuf, cap_len, t);
    // repack (updated) h; reused by preds now and by next step's GEMMs
    pack_a_k<<<cdiv(B_ * D_, BS), BS, 0, stream>>>(hBuf, pH, B_, D_);
    // preds[:, t, :] = mask ? h @ W_fc + b_fc : 0   (32x512 @ 512x20000)
    gemm_preds_k<<<cdiv(2 * (V_ / 16), 8), BS, 0, stream>>>(pH, pWfc, b_fc, cap_len, out_preds, t);
  }
}
